// DenseMoELayer_10342281249413
// MI455X (gfx1250) — compile-verified, matrix operand order checked by reading, not run on hardware
//
#include <hip/hip_runtime.h>
#include <hip/hip_fp16.h>
#include <math.h>
#include <stdint.h>

typedef __attribute__((ext_vector_type(16))) _Float16 v16h;
typedef __attribute__((ext_vector_type(8)))  _Float16 v8h;
typedef __attribute__((ext_vector_type(8)))  float    v8f;
typedef __attribute__((ext_vector_type(4)))  float    v4f;
typedef __attribute__((ext_vector_type(4)))  unsigned int u32x4;
typedef __attribute__((ext_vector_type(8)))  int i32x8;
typedef __attribute__((ext_vector_type(4)))  int i32x4;

#if defined(__has_builtin)
#if __has_builtin(__builtin_amdgcn_tensor_load_to_lds) && \
    __has_builtin(__builtin_amdgcn_s_wait_tensorcnt)
#define HAVE_TDM 1
#endif
#endif

// Problem constants
constexpr int T_DIM = 4096;
constexpr int D_DIM = 1024;
constexpr int F_DIM = 2048;
constexpr int E_DIM = 8;

// Tiling: block 128(M) x 128(N) x 32(K); 8 waves, wave tile 32(M) x 64(N).
constexpr int BM = 128;
constexpr int BN = 128;
constexpr int BK = 32;
constexpr int TILE = BM * BK;   // halves per (square-ish) LDS tile buffer

// ---------------------------------------------------------------------------
// WMMA helper: D = A * B + C   (16x16x32, f16 in, f32 accumulate)
// ---------------------------------------------------------------------------
__device__ __forceinline__ v8f wmma_f16(v16h a, v16h b, v8f c) {
  return __builtin_amdgcn_wmma_f32_16x16x32_f16(
      false, a, false, b, (short)0, c, false, false);
}

// Build a 16x32 f16 fragment from a row-major LDS tile (row stride = BK).
// lane<16: row=lane, K in {0..7,16..23}; lane>=16: row=lane-16, K in {8..15,24..31}.
__device__ __forceinline__ v16h load_frag(const _Float16* lds, int row, int fh) {
  const _Float16* p = lds + row * BK;
  v8h lo = *(const v8h*)(p + fh * 8);
  v8h hi = *(const v8h*)(p + 16 + fh * 8);
  return __builtin_shufflevector(lo, hi, 0, 1, 2, 3, 4, 5, 6, 7,
                                 8, 9, 10, 11, 12, 13, 14, 15);
}

__device__ __forceinline__ v8h cvt8(v4f a, v4f b) {
  v8h r;
  r[0] = (_Float16)a[0]; r[1] = (_Float16)a[1];
  r[2] = (_Float16)a[2]; r[3] = (_Float16)a[3];
  r[4] = (_Float16)b[0]; r[5] = (_Float16)b[1];
  r[6] = (_Float16)b[2]; r[7] = (_Float16)b[3];
  return r;
}

// 16 contiguous floats <-> 4 v4f registers; f16 pack store (2x16B) to LDS.
__device__ __forceinline__ void ld16(const float* p, v4f r[4]) {
  r[0] = *(const v4f*)(p);
  r[1] = *(const v4f*)(p + 4);
  r[2] = *(const v4f*)(p + 8);
  r[3] = *(const v4f*)(p + 12);
}
__device__ __forceinline__ void st16(_Float16* d, const v4f r[4]) {
  *(v8h*)(d)     = cvt8(r[0], r[1]);
  *(v8h*)(d + 8) = cvt8(r[2], r[3]);
}

// ---------------------------------------------------------------------------
// Kernel 1: fused top-2 gating -> dense [T,E] weights.
// ---------------------------------------------------------------------------
__global__ __launch_bounds__(256) void moe_gating_kernel(
    const float* __restrict__ gating, float* __restrict__ w_full) {
  int t = blockIdx.x * blockDim.x + threadIdx.x;
  if (t >= T_DIM) return;
  float l[E_DIM];
  float m = -INFINITY;
#pragma unroll
  for (int e = 0; e < E_DIM; ++e) {
    l[e] = gating[t * E_DIM + e];
    m = fmaxf(m, l[e]);
  }
  float p[E_DIM];
#pragma unroll
  for (int e = 0; e < E_DIM; ++e) p[e] = __expf(l[e] - m);
  int i1 = 0;
#pragma unroll
  for (int e = 1; e < E_DIM; ++e) if (p[e] > p[i1]) i1 = e;
  int i2 = (i1 == 0) ? 1 : 0;
#pragma unroll
  for (int e = 0; e < E_DIM; ++e) if (e != i1 && p[e] > p[i2]) i2 = e;
  float rden = __builtin_amdgcn_rcpf(p[i1] + p[i2]);
#pragma unroll
  for (int e = 0; e < E_DIM; ++e) {
    float w = 0.0f;
    if (e == i1) w = p[i1] * rden;
    if (e == i2) w = p[i2] * rden;
    w_full[t * E_DIM + e] = w;
  }
}

// ---------------------------------------------------------------------------
// Kernel 2: per-expert fused gate+up GEMM + SiLU-gate epilogue -> h (f16).
// Double-buffered LDS, one barrier per K-step: compute from buffer `cur`
// while register-prefetched f32 slices are converted into buffer `cur^1`.
// ---------------------------------------------------------------------------
__global__ __launch_bounds__(256) void moe_gateup_kernel(
    const float* __restrict__ x,   // [T, D]
    const float* __restrict__ gw,  // [F, D]
    const float* __restrict__ uw,  // [F, D]
    _Float16* __restrict__ h) {    // [T, F]
  __shared__ _Float16 sA[2 * TILE];  // 16 KB
  __shared__ _Float16 sG[2 * TILE];  // 16 KB
  __shared__ _Float16 sU[2 * TILE];  // 16 KB

  const int tid = threadIdx.x;
  const int lane = tid & 31;
  const int wave = tid >> 5;
  const int waveM = wave >> 1;   // 0..3  (32 rows each)
  const int waveN = wave & 1;    // 0..1  (64 cols each)
  const int fr = lane & 15;
  const int fh = lane >> 4;

  const int blockM = blockIdx.x * BM;
  const int blockN = blockIdx.y * BN;

  // Global->LDS mapping: 128 rows x 32 cols, 2 threads/row, 16 floats each.
  const int row = tid >> 1, col = (tid & 1) * 16;
  const int soff = row * BK + col;
  const float* pax = x  + (size_t)(blockM + row) * D_DIM + col;
  const float* pgw = gw + (size_t)(blockN + row) * D_DIM + col;
  const float* puw = uw + (size_t)(blockN + row) * D_DIM + col;

  v4f ra[4], rg[4], ru[4];
  ld16(pax, ra);
  ld16(pgw, rg);
  ld16(puw, ru);
  st16(&sA[soff], ra);
  st16(&sG[soff], rg);
  st16(&sU[soff], ru);
  __syncthreads();

  v8f cg[2][4] = {};
  v8f cu[2][4] = {};

  for (int k0 = 0; k0 < D_DIM; k0 += BK) {
    const int cur = (k0 / BK) & 1;
    const int nxt = cur ^ 1;
    const bool has_next = (k0 + BK) < D_DIM;

    if (has_next) {  // issue next-slice global loads under the WMMA chain
      ld16(pax + k0 + BK, ra);
      ld16(pgw + k0 + BK, rg);
      ld16(puw + k0 + BK, ru);
    }

    const _Float16* A = sA + cur * TILE;
    const _Float16* G = sG + cur * TILE;
    const _Float16* U = sU + cur * TILE;
    v16h aF[2], gF[4], uF[4];
#pragma unroll
    for (int i = 0; i < 2; ++i)
      aF[i] = load_frag(A, waveM * 32 + i * 16 + fr, fh);
#pragma unroll
    for (int j = 0; j < 4; ++j) {
      gF[j] = load_frag(G, waveN * 64 + j * 16 + fr, fh);
      uF[j] = load_frag(U, waveN * 64 + j * 16 + fr, fh);
    }
#pragma unroll
    for (int i = 0; i < 2; ++i)
#pragma unroll
      for (int j = 0; j < 4; ++j) {
        cg[i][j] = wmma_f16(aF[i], gF[j], cg[i][j]);
        cu[i][j] = wmma_f16(aF[i], uF[j], cu[i][j]);
      }

    if (has_next) {  // fill the other buffer; published by the barrier below
      st16(&sA[nxt * TILE + soff], ra);
      st16(&sG[nxt * TILE + soff], rg);
      st16(&sU[nxt * TILE + soff], ru);
    }
    __syncthreads();
  }

  // Epilogue: silu(g)*u -> f16.  C layout: n = lane&15, m = r + 8*(lane>>4).
#pragma unroll
  for (int i = 0; i < 2; ++i)
#pragma unroll
    for (int j = 0; j < 4; ++j) {
      const int m_base = blockM + waveM * 32 + i * 16 + fh * 8;
      const int n = blockN + waveN * 64 + j * 16 + fr;
#pragma unroll
      for (int r = 0; r < 8; ++r) {
        float g = cg[i][j][r];
        float u = cu[i][j][r];
        float sig = __builtin_amdgcn_rcpf(1.0f + __expf(-g));
        h[(size_t)(m_base + r) * F_DIM + n] = (_Float16)(g * sig * u);
      }
    }
}

// ---------------------------------------------------------------------------
// Kernel 3: per-expert down GEMM + gating-weight epilogue.
// A-operand (h, f16) pipelined into double-buffered LDS by the Tensor Data
// Mover (issue for k+1 at top of step k, wait only after the WMMA chain);
// B-operand converted f32->f16 by the threads while the TDM runs.
// ---------------------------------------------------------------------------
__global__ __launch_bounds__(256) void moe_down_kernel(
    const _Float16* __restrict__ h,  // [T, F] f16
    const float* __restrict__ dw,    // [D, F]
    const float* __restrict__ wcol,  // w_full + e (stride E)
    float* __restrict__ out,         // [T, D]
    int accum) {
  __shared__ _Float16 sA[2 * TILE];  // 16 KB (TDM destination, ping-pong)
  __shared__ _Float16 sB[2 * TILE];  // 16 KB

  const int tid = threadIdx.x;
  const int lane = tid & 31;
  const int wave = tid >> 5;
  const int waveM = wave >> 1;
  const int waveN = wave & 1;
  const int fr = lane & 15;
  const int fh = lane >> 4;

  const int blockM = blockIdx.x * BM;  // tokens
  const int blockN = blockIdx.y * BN;  // hidden dim D

  const int row = tid >> 1, col = (tid & 1) * 16;
  const int soff = row * BK + col;
  const float* pb = dw + (size_t)(blockN + row) * F_DIM + col;

#if HAVE_TDM
  // Tensor DMA descriptor (2D): tensor [T_DIM, F_DIM] f16, tile [BM, BK].
  const uint32_t lds_a = (uint32_t)(uintptr_t)(void*)sA;
  const uint64_t gbase = (uint64_t)(uintptr_t)(h + (size_t)blockM * F_DIM);
  i32x8 g1;
  g1[0] = (int)(1u << 16);                                  // data_size = 2B
  g1[1] = (int)(((uint32_t)F_DIM & 0xFFFFu) << 16);         // tensor_dim0 lo
  g1[2] = (int)(((uint32_t)F_DIM >> 16) |
                (((uint32_t)T_DIM & 0xFFFFu) << 16));       // dim0 hi | dim1 lo
  g1[3] = (int)(((uint32_t)T_DIM >> 16) |
                ((uint32_t)BK << 16));                      // dim1 hi | tile_dim0
  g1[4] = (int)(uint32_t)BM;                                // tile_dim1 (tile_dim2=0)
  g1[5] = (int)(uint32_t)F_DIM;                             // tensor_dim0_stride lo
  g1[6] = 0;
  g1[7] = 0;
  const i32x4 z4 = {0, 0, 0, 0};
  const i32x8 z8 = {0, 0, 0, 0, 0, 0, 0, 0};
#define TDM_ISSUE(K0, BUF)                                                    \
  do {                                                                        \
    uint64_t ga = gbase + (uint64_t)(K0) * sizeof(_Float16);                  \
    u32x4 g0;                                                                 \
    g0[0] = 1u;                                                               \
    g0[1] = lds_a + (uint32_t)((BUF) * TILE * sizeof(_Float16));              \
    g0[2] = (uint32_t)ga;                                                     \
    g0[3] = (uint32_t)((ga >> 32) & 0x01FFFFFFu) | (2u << 30);                \
    __builtin_amdgcn_tensor_load_to_lds(g0, g1, z4, z4, z8, 0);               \
  } while (0)
#else
  const _Float16* pa = h + (size_t)(blockM + row) * F_DIM + col;
#endif

  // ---- prologue: fill buffer 0 ----
  v4f rb[4];
  ld16(pb, rb);
#if HAVE_TDM
  if (wave == 0) TDM_ISSUE(0, 0);
#else
  *(v8h*)&sA[soff]     = *(const v8h*)(pa);
  *(v8h*)&sA[soff + 8] = *(const v8h*)(pa + 8);
#endif
  st16(&sB[soff], rb);
  ld16(pb + BK, rb);  // slice k=BK into regs
#if HAVE_TDM
  if (wave == 0) __builtin_amdgcn_s_wait_tensorcnt(0);
#endif
  __syncthreads();

  v8f c[2][4] = {};

  for (int k0 = 0; k0 < F_DIM; k0 += BK) {
    const int cur = (k0 / BK) & 1;
    const int nxt = cur ^ 1;
    const bool has_next = (k0 + BK) < F_DIM;

#if HAVE_TDM
    if (has_next && wave == 0) TDM_ISSUE(k0 + BK, nxt);  // overlaps compute
#else
    if (has_next) {
      *(v8h*)&sA[nxt * TILE + soff]     = *(const v8h*)(pa + k0 + BK);
      *(v8h*)&sA[nxt * TILE + soff + 8] = *(const v8h*)(pa + k0 + BK + 8);
    }
#endif

    const _Float16* A = sA + cur * TILE;
    const _Float16* B = sB + cur * TILE;
    v16h aF[2], bF[4];
#pragma unroll
    for (int i = 0; i < 2; ++i)
      aF[i] = load_frag(A, waveM * 32 + i * 16 + fr, fh);
#pragma unroll
    for (int j = 0; j < 4; ++j)
      bF[j] = load_frag(B, waveN * 64 + j * 16 + fr, fh);
#pragma unroll
    for (int i = 0; i < 2; ++i)
#pragma unroll
      for (int j = 0; j < 4; ++j)
        c[i][j] = wmma_f16(aF[i], bF[j], c[i][j]);

    if (has_next) {
      st16(&sB[nxt * TILE + soff], rb);                       // publish B k+1
      if (k0 + 2 * BK < F_DIM) ld16(pb + k0 + 2 * BK, rb);    // fetch B k+2
#if HAVE_TDM
      if (wave == 0) __builtin_amdgcn_s_wait_tensorcnt(0);    // TDM k+1 done
#endif
    }
    __syncthreads();
  }

#pragma unroll
  for (int i = 0; i < 2; ++i)
#pragma unroll
    for (int j = 0; j < 4; ++j) {
      const int m_base = blockM + waveM * 32 + i * 16 + fh * 8;
      const int n = blockN + waveN * 64 + j * 16 + fr;
#pragma unroll
      for (int r = 0; r < 8; ++r) {
        const int t = m_base + r;
        float wgt = wcol[(size_t)t * E_DIM];
        float val = wgt * c[i][j][r];
        size_t o = (size_t)t * D_DIM + n;
        out[o] = accum ? (out[o] + val) : val;
      }
    }
}

// ---------------------------------------------------------------------------
// Launch: gating, then per expert: gate+up -> down (sequential on stream).
// Workspace: h (T*F f16 = 16 MB) then w_full (T*E f32).
// ---------------------------------------------------------------------------
extern "C" void kernel_launch(void* const* d_in, const int* in_sizes, int n_in,
                              void* d_out, int out_size, void* d_ws, size_t ws_size,
                              hipStream_t stream) {
  const float* x      = (const float*)d_in[0];
  const float* gating = (const float*)d_in[1];
  const float* gate_w = (const float*)d_in[2];
  const float* up_w   = (const float*)d_in[3];
  const float* down_w = (const float*)d_in[4];
  float* out = (float*)d_out;

  _Float16* h_ws = (_Float16*)d_ws;
  float* w_full = (float*)((char*)d_ws + (size_t)T_DIM * F_DIM * sizeof(_Float16));

  moe_gating_kernel<<<T_DIM / 256, 256, 0, stream>>>(gating, w_full);

  for (int e = 0; e < E_DIM; ++e) {
    const size_t woff = (size_t)e * F_DIM * D_DIM;
    moe_gateup_kernel<<<dim3(T_DIM / BM, F_DIM / BN), 256, 0, stream>>>(
        x, gate_w + woff, up_w + woff, h_ws);
    moe_down_kernel<<<dim3(T_DIM / BM, D_DIM / BN), 256, 0, stream>>>(
        h_ws, down_w + woff, w_full + e, out, e == 0 ? 0 : 1);
  }
}
